// GRU_55164559949998
// MI455X (gfx1250) — matrix operand order repaired and checked by reading, not compiled, not run on hardware
//
#include <hip/hip_runtime.h>
#include <hip/hip_bf16.h>
#include <math.h>

// ---------------------------------------------------------------------------
// GRU on gfx1250: persistent batch-tiled recurrence using v_wmma_f32_16x16x32_bf16
//   L=512, B=256, DIN=256, DL=256, K = DL+DIN = 512
//   grid: 16 blocks (one per 16-row batch tile) x 256 threads (8 wave32)
//   each wave owns 32 output columns (2 N-tiles of 16)
// ---------------------------------------------------------------------------

typedef __bf16 v16bf __attribute__((ext_vector_type(16)));
typedef __bf16 v8bf  __attribute__((ext_vector_type(8)));
typedef float  v8f   __attribute__((ext_vector_type(8)));

// addrspace(1) (global) pointer for weight-fragment loads -> global_load_b128
typedef const v16bf __attribute__((address_space(1)))* gv16p;

#define L_SEQ 512
#define BATCH 256
#define DIN_  256
#define DL_   256
#define KTOT  512          // DL + DIN
#define LDP   264          // padded LDS row pitch in bf16 elems (528B = 33*16B)

// ---------------------------------------------------------------------------
// Kernel 0: pack fp32 weights [512,256] (Wz,Wr,Wh) into the WMMA B-fragment
// layout, bf16, in workspace. B (32x16, 16-bit): lanes 0-15 hold K=0..7,16..23 of
// column N=lane, lanes 16-31 hold K=8..15,24..31 of column N=lane-16; 16 bf16/lane.
// Flat dst index == flat thread id by construction.
// ---------------------------------------------------------------------------
__global__ __launch_bounds__(256) void gru_pack_w(const float* __restrict__ Wz,
                                                  const float* __restrict__ Wr,
                                                  const float* __restrict__ Wh,
                                                  __bf16* __restrict__ wsW) {
  int t = blockIdx.x * 256 + threadIdx.x;   // 3*16*16*32*16 = 393216 total
  int j    =  t        & 15;                // element within lane's fragment
  int lane = (t >> 4)  & 31;
  int nt   = (t >> 9)  & 15;                // N-tile (16 cols each)
  int kt   = (t >> 13) & 15;                // K-tile (32 rows each)
  int g    =  t >> 17;                      // 0=z, 1=r, 2=h(cand)
  const float* W = (g == 0) ? Wz : (g == 1) ? Wr : Wh;
  int k = kt * 32 + ((lane >> 4) << 3) + (j & 7) + ((j >> 3) << 4); // 0..511
  int n = nt * 16 + (lane & 15);            // 0..255
  wsW[t] = (__bf16)W[k * DL_ + n];
}

// ---------------------------------------------------------------------------
// helpers
// ---------------------------------------------------------------------------
__device__ __forceinline__ v16bf load_a_frag(const __bf16* rowbase, int col) {
  // A (16x32, 16-bit): lane l holds row M=l%16; K-octets at col and col+16
  v8bf c0 = *(const v8bf*)(rowbase + col);
  v8bf c1 = *(const v8bf*)(rowbase + col + 16);
  v16bf a;
#pragma unroll
  for (int i = 0; i < 8; ++i) { a[i] = c0[i]; a[i + 8] = c1[i]; }
  return a;
}

// Weight base laundered as a scalar integer each timestep (defeats LICM that
// would otherwise keep 96 B-fragments live across the t-loop and spill), then
// rebuilt as an explicit addrspace(1) pointer so loads are global_load_b128
// (LOADcnt only), not flat_load (LOADcnt+DScnt).
__device__ __forceinline__ unsigned long long launder_base(const void* p) {
  unsigned long long v = (unsigned long long)p;
  asm volatile("" : "+s"(v));
  return v;
}

__device__ __forceinline__ v16bf load_b_frag(unsigned long long Wbase,
                                             int g, int kt, int nt, int lane) {
  // element index * 2 bytes
  unsigned long long elem =
      ((((unsigned long long)((g * 16 + kt) * 16 + nt)) * 32 + lane) * 16);
  return *(gv16p)(Wbase + elem * 2ull);   // 32B contiguous per lane
}

__device__ __forceinline__ v8f splat8(float x) {
  v8f v;
#pragma unroll
  for (int i = 0; i < 8; ++i) v[i] = x;
  return v;
}

__device__ __forceinline__ float sigmoid_f(float x) {
  return 1.0f / (1.0f + __expf(-x));
}
__device__ __forceinline__ float tanh_f(float x) {
  return 2.0f / (1.0f + __expf(-2.0f * x)) - 1.0f;
}

#define WMMA_BF16(A, B, C) \
  __builtin_amdgcn_wmma_f32_16x16x32_bf16(false, (A), false, (B), (short)0, (C), false, false)

// ---------------------------------------------------------------------------
// Kernel 1: persistent sequential GRU. One block per 16 batch rows.
// ---------------------------------------------------------------------------
__global__ __launch_bounds__(256) void gru_seq(const float* __restrict__ X,
                                               const float* __restrict__ bz,
                                               const float* __restrict__ br,
                                               const float* __restrict__ bh,
                                               const __bf16* __restrict__ wsW,
                                               float* __restrict__ H,
                                               float* __restrict__ Z,
                                               float* __restrict__ R) {
  __shared__ __align__(16) __bf16 hbuf [16 * LDP];  // h      (bf16 copy)
  __shared__ __align__(16) __bf16 rhbuf[16 * LDP];  // r * h
  __shared__ __align__(16) __bf16 xbuf [16 * LDP];  // x[t] tile

  const int tid  = threadIdx.x;
  const int lane = tid & 31;
  const int w    = tid >> 5;            // wave id 0..7
  const int bt   = blockIdx.x * 16;     // batch-row base of this tile

  // ---- init: h = 0 (LDS) and H[0] = 0 (global) --------------------------
  for (int i = tid; i < 16 * LDP; i += 256) hbuf[i] = (__bf16)0.0f;
  for (int i = tid; i < 16 * DL_; i += 256) {
    int m = i >> 8, n = i & 255;
    H[(size_t)(bt + m) * DL_ + n] = 0.0f;
  }

  // per-lane bias values for this wave's two N-tiles
  const int nt0 = w * 2, nt1 = w * 2 + 1;
  const int nA = nt0 * 16 + (lane & 15);
  const int nB = nt1 * 16 + (lane & 15);
  const float bzA = bz[nA], bzB = bz[nB];
  const float brA = br[nA], brB = br[nB];
  const float bhA = bh[nA], bhB = bh[nB];

  const int rowA  = lane & 15;          // A-fragment row (M)
  const int koff  = (lane >> 4) << 3;   // A-fragment K-octet select
  const int m0    = (lane >> 4) << 3;   // C-fragment row base
  const __bf16* arow_h  = hbuf  + rowA * LDP;
  const __bf16* arow_rh = rhbuf + rowA * LDP;
  const __bf16* arow_x  = xbuf  + rowA * LDP;

  __syncthreads();

#pragma unroll 1
  for (int t = 0; t < L_SEQ; ++t) {
    // loop-variant scalar base of the packed weights (see launder comment)
    const unsigned long long Wv = launder_base(wsW);

    // ---- stage x[t] tile into LDS as bf16 (vectorized) ------------------
    {
      const int m = tid >> 4, seg = tid & 15;          // 16 consecutive cols
      const float4* xr =
          (const float4*)(X + ((size_t)t * BATCH + bt + m) * DIN_ + seg * 16);
      float4 f0 = xr[0], f1 = xr[1], f2 = xr[2], f3 = xr[3];
      v8bf o0, o1;
      o0[0] = (__bf16)f0.x; o0[1] = (__bf16)f0.y; o0[2] = (__bf16)f0.z; o0[3] = (__bf16)f0.w;
      o0[4] = (__bf16)f1.x; o0[5] = (__bf16)f1.y; o0[6] = (__bf16)f1.z; o0[7] = (__bf16)f1.w;
      o1[0] = (__bf16)f2.x; o1[1] = (__bf16)f2.y; o1[2] = (__bf16)f2.z; o1[3] = (__bf16)f2.w;
      o1[4] = (__bf16)f3.x; o1[5] = (__bf16)f3.y; o1[6] = (__bf16)f3.z; o1[7] = (__bf16)f3.w;
      *(v8bf*)&xbuf[m * LDP + seg * 16]     = o0;      // two 16B ds stores
      *(v8bf*)&xbuf[m * LDP + seg * 16 + 8] = o1;
      if (t + 1 < L_SEQ)  // prefetch next timestep's row into L2
        __builtin_prefetch((const float*)xr + (size_t)BATCH * DIN_, 0, 1);
    }
    __syncthreads();

    // ---- gates: z = sigm([h|x]Wz+bz), r = sigm([h|x]Wr+br) --------------
    v8f zA = splat8(bzA), zB = splat8(bzB);
    v8f rA = splat8(brA), rB = splat8(brB);
#pragma unroll 2
    for (int kt = 0; kt < 8; ++kt) {          // K over h
      v16bf a = load_a_frag(arow_h, kt * 32 + koff);
      zA = WMMA_BF16(a, load_b_frag(Wv, 0, kt, nt0, lane), zA);
      zB = WMMA_BF16(a, load_b_frag(Wv, 0, kt, nt1, lane), zB);
      rA = WMMA_BF16(a, load_b_frag(Wv, 1, kt, nt0, lane), rA);
      rB = WMMA_BF16(a, load_b_frag(Wv, 1, kt, nt1, lane), rB);
    }
#pragma unroll 2
    for (int kt = 8; kt < 16; ++kt) {         // K over x
      v16bf a = load_a_frag(arow_x, (kt - 8) * 32 + koff);
      zA = WMMA_BF16(a, load_b_frag(Wv, 0, kt, nt0, lane), zA);
      zB = WMMA_BF16(a, load_b_frag(Wv, 0, kt, nt1, lane), zB);
      rA = WMMA_BF16(a, load_b_frag(Wv, 1, kt, nt0, lane), rA);
      rB = WMMA_BF16(a, load_b_frag(Wv, 1, kt, nt1, lane), rB);
    }

#pragma unroll
    for (int v = 0; v < 8; ++v) {
      const int m = m0 + v;
      zA[v] = sigmoid_f(zA[v]);  zB[v] = sigmoid_f(zB[v]);
      rA[v] = sigmoid_f(rA[v]);  rB[v] = sigmoid_f(rB[v]);
      const float hvA = (float)hbuf[m * LDP + nA];
      const float hvB = (float)hbuf[m * LDP + nB];
      const size_t o = ((size_t)t * BATCH + bt + m) * DL_;
      Z[o + nA] = zA[v];  Z[o + nB] = zB[v];
      R[o + nA] = rA[v];  R[o + nB] = rB[v];
      rhbuf[m * LDP + nA] = (__bf16)(rA[v] * hvA);
      rhbuf[m * LDP + nB] = (__bf16)(rB[v] * hvB);
    }
    __syncthreads();   // rhbuf ready; all hbuf A-reads done

    // ---- candidate: tanh([r*h|x]Wh + bh) --------------------------------
    v8f cA = splat8(bhA), cB = splat8(bhB);
#pragma unroll 2
    for (int kt = 0; kt < 8; ++kt) {          // K over r*h
      v16bf a = load_a_frag(arow_rh, kt * 32 + koff);
      cA = WMMA_BF16(a, load_b_frag(Wv, 2, kt, nt0, lane), cA);
      cB = WMMA_BF16(a, load_b_frag(Wv, 2, kt, nt1, lane), cB);
    }
#pragma unroll 2
    for (int kt = 8; kt < 16; ++kt) {         // K over x
      v16bf a = load_a_frag(arow_x, (kt - 8) * 32 + koff);
      cA = WMMA_BF16(a, load_b_frag(Wv, 2, kt, nt0, lane), cA);
      cB = WMMA_BF16(a, load_b_frag(Wv, 2, kt, nt1, lane), cB);
    }

    // ---- h_new = (1-z)h + z*cand; write H[t+1], update LDS h ------------
#pragma unroll
    for (int v = 0; v < 8; ++v) {
      const int m = m0 + v;
      const float hvA = (float)hbuf[m * LDP + nA];   // re-read (own cols)
      const float hvB = (float)hbuf[m * LDP + nB];
      const float cdA = tanh_f(cA[v]);
      const float cdB = tanh_f(cB[v]);
      const float hnA = (1.0f - zA[v]) * hvA + zA[v] * cdA;
      const float hnB = (1.0f - zB[v]) * hvB + zB[v] * cdB;
      hbuf[m * LDP + nA] = (__bf16)hnA;
      hbuf[m * LDP + nB] = (__bf16)hnB;
      const size_t o = ((size_t)(t + 1) * BATCH + bt + m) * DL_;
      H[o + nA] = hnA;  H[o + nB] = hnB;
    }
    __syncthreads();   // hbuf update visible before next step's reads
  }
}

// ---------------------------------------------------------------------------
// launch
// ---------------------------------------------------------------------------
extern "C" void kernel_launch(void* const* d_in, const int* in_sizes, int n_in,
                              void* d_out, int out_size, void* d_ws, size_t ws_size,
                              hipStream_t stream) {
  const float* X  = (const float*)d_in[0];
  const float* Wz = (const float*)d_in[1];
  const float* bz = (const float*)d_in[2];
  const float* Wr = (const float*)d_in[3];
  const float* br = (const float*)d_in[4];
  const float* Wh = (const float*)d_in[5];
  const float* bh = (const float*)d_in[6];

  __bf16* wsW = (__bf16*)d_ws;   // 3*512*256 bf16 = 768 KB packed weights

  float* H = (float*)d_out;                                  // [513,256,256]
  float* Z = H + (size_t)(L_SEQ + 1) * BATCH * DL_;          // [512,256,256]
  float* R = Z + (size_t)L_SEQ * BATCH * DL_;                // [512,256,256]

  // pack weights into WMMA B-fragment layout (bf16)
  gru_pack_w<<<(3 * KTOT * DL_) / 256, 256, 0, stream>>>(Wz, Wr, Wh, wsW);

  // persistent sequential GRU: one block per 16 batch rows
  gru_seq<<<BATCH / 16, 256, 0, stream>>>(X, bz, br, bh, wsW, H, Z, R);
}
